// Block_18846316495189
// MI455X (gfx1250) — matrix operand order repaired and checked by reading, not compiled, not run on hardware
//
#include <hip/hip_runtime.h>
#include <cmath>

// ---------------------------------------------------------------------------
// Types for CDNA5 WMMA (wave32): V_WMMA_F32_16X16X32_F16
// ---------------------------------------------------------------------------
typedef __attribute__((ext_vector_type(16))) _Float16 v16h;
typedef __attribute__((ext_vector_type(8)))  _Float16 v8h;
typedef __attribute__((ext_vector_type(8)))  float    v8f;

union AF { v16h v; v8h h[2]; };

__device__ __forceinline__ v8h zero8h() {
  v8h z;
#pragma unroll
  for (int i = 0; i < 8; ++i) z[i] = (_Float16)0;
  return z;
}
__device__ __forceinline__ v8f zero8f() {
  v8f z;
#pragma unroll
  for (int i = 0; i < 8; ++i) z[i] = 0.0f;
  return z;
}

// D = A(16x32 f16) * B(32x16 f16) + C(16x16 f32)
__device__ __forceinline__ v8f wmma16x16x32(v16h a, v16h b, v8f c) {
  return __builtin_amdgcn_wmma_f32_16x16x32_f16(false, a, false, b, (short)0, c,
                                                false, false);
}

// Same-wave LDS staging fence: DS ops from one wave complete in order at the
// LDS unit (ISA 7.3), so only compiler reordering must be prevented.
__device__ __forceinline__ void lds_fence() {
  asm volatile("" ::: "memory");
  __builtin_amdgcn_wave_barrier();
  asm volatile("" ::: "memory");
}

// Reduction across the 16 lanes of a wave32 half (C/D rows live across lanes)
__device__ __forceinline__ float redadd16(float v) {
  v += __shfl_xor(v, 1);
  v += __shfl_xor(v, 2);
  v += __shfl_xor(v, 4);
  v += __shfl_xor(v, 8);
  return v;
}
// Full-wave max (once per 32-key block in attention)
__device__ __forceinline__ float redmax32(float v) {
  v = fmaxf(v, __shfl_xor(v, 1));
  v = fmaxf(v, __shfl_xor(v, 2));
  v = fmaxf(v, __shfl_xor(v, 4));
  v = fmaxf(v, __shfl_xor(v, 8));
  v = fmaxf(v, __shfl_xor(v, 16));
  return v;
}

constexpr int DM  = 64;    // d_model
constexpr int NH  = 6;     // heads
constexpr int HS  = 10;    // head size
constexpr int HSP = 16;    // padded head size
constexpr int TT  = 2048;  // seq len
constexpr int BB  = 8;     // batch
constexpr int DFF = 256;   // ffn dim

// ---------------------------------------------------------------------------
// K1: per-head QKV projection as WMMA GEMM (16-token tile per wave).
//   Q[b,h][t][16] f16, pre-scaled by 1/sqrt(HS), cols 10..15 zero
//   K[b,h][t][16] f16, cols 10..15 zero
//   Vt[b,h][16][t] f16 transposed; row 10 = 1.0 (softmax-denominator column)
// ---------------------------------------------------------------------------
__global__ __launch_bounds__(256) void qkv_kernel(
    const float* __restrict__ x, const float* __restrict__ Wqkv,
    _Float16* __restrict__ Q, _Float16* __restrict__ K,
    _Float16* __restrict__ Vt) {
  __shared__ __align__(16) _Float16 wq[NH * 32 * DM];  // [h][e:32][k:64], 24 KB
  for (int i = threadIdx.x; i < NH * 32 * DM; i += 256) {
    int h = i >> 11;
    int rem = i & 2047;
    int e = rem >> 6;
    int k = rem & 63;
    wq[i] = (e < 3 * HS) ? (_Float16)Wqkv[(h * DM + k) * (3 * HS) + e]
                         : (_Float16)0;
  }
  __syncthreads();

  int wave = threadIdx.x >> 5, lane = threadIdx.x & 31;
  int idx = lane & 15, hi = lane >> 4;
  int row0 = (blockIdx.x * 8 + wave) * 16;
  int b = row0 / TT;
  int t0 = row0 - b * TT;

  const float* xr = x + (size_t)(row0 + idx) * DM;
  AF a0, a1;
#pragma unroll
  for (int j = 0; j < 8; ++j) {
    a0.h[0][j] = (_Float16)xr[hi * 16 + j];
    a0.h[1][j] = (_Float16)xr[hi * 16 + 8 + j];
    a1.h[0][j] = (_Float16)xr[32 + hi * 16 + j];
    a1.h[1][j] = (_Float16)xr[32 + hi * 16 + 8 + j];
  }

  const float qs = 0.3162277660168379f;  // 1/sqrt(10)
#pragma unroll
  for (int h = 0; h < NH; ++h) {
    v8f acc[2];
#pragma unroll
    for (int nt = 0; nt < 2; ++nt) {
      AF b0, b1v;
      const _Float16* wr = wq + ((size_t)h * 32 + nt * 16 + idx) * DM;
      b0.h[0]  = *(const v8h*)(wr + hi * 16);
      b0.h[1]  = *(const v8h*)(wr + hi * 16 + 8);
      b1v.h[0] = *(const v8h*)(wr + 32 + hi * 16);
      b1v.h[1] = *(const v8h*)(wr + 32 + hi * 16 + 8);
      v8f c = wmma16x16x32(a0.v, b0.v, zero8f());
      acc[nt] = wmma16x16x32(a1.v, b1v.v, c);
    }
    size_t qbase = ((size_t)b * NH + h) * TT;
    _Float16* vtb = Vt + ((size_t)b * NH + h) * HSP * TT;
#pragma unroll
    for (int nt = 0; nt < 2; ++nt) {
      int col = nt * 16 + idx;  // [q(10) | k(10) | v(10) | pad(2)]
#pragma unroll
      for (int r = 0; r < 8; ++r) {
        int tok = t0 + hi * 8 + r;
        float v = acc[nt][r];
        if (col < HS) {
          Q[(qbase + tok) * HSP + col] = (_Float16)(v * qs);
        } else if (col < 2 * HS) {
          K[(qbase + tok) * HSP + (col - HS)] = (_Float16)v;
        } else if (col < 3 * HS) {
          vtb[(size_t)(col - 2 * HS) * TT + tok] = (_Float16)v;
        }
        if (nt == 1 && idx >= HS) {  // 6 lanes emit the padding lanes
          Q[(qbase + tok) * HSP + idx] = (_Float16)0;
          K[(qbase + tok) * HSP + idx] = (_Float16)0;
          vtb[(size_t)idx * TT + tok] = (idx == HS) ? (_Float16)1 : (_Float16)0;
        }
      }
    }
  }
}

// ---------------------------------------------------------------------------
// K2: causal flash attention, one wave32 per 16-row query tile.
// Tile-level running max (softmax exactly invariant to per-row shifts) and
// the row-sum accumulated by WMMA through the ones-column (Vt row 10).
// ---------------------------------------------------------------------------
template <bool MASK>
__device__ __forceinline__ void attn_block(
    int base, int ibase, int idx, int hi, const _Float16* __restrict__ Kb,
    const _Float16* __restrict__ Vb, const AF& qa, _Float16* lds_p, v8f& O,
    float& mtile) {
  // K fragments (B 32x16): lane n=idx, k=hi*16+i; k>=16 is zero head padding.
  AF kf0, kf1;
  if (hi == 0) {
    const _Float16* kr0 = Kb + (size_t)(base + idx) * HSP;
    const _Float16* kr1 = Kb + (size_t)(base + 16 + idx) * HSP;
    kf0.h[0] = *(const v8h*)(kr0);
    kf0.h[1] = *(const v8h*)(kr0 + 8);
    kf1.h[0] = *(const v8h*)(kr1);
    kf1.h[1] = *(const v8h*)(kr1 + 8);
  } else {
    kf0.h[0] = zero8h(); kf0.h[1] = zero8h();
    kf1.h[0] = zero8h(); kf1.h[1] = zero8h();
  }
  // V fragment loaded early to overlap with softmax VALU work.
  AF vf;
  const _Float16* vr = Vb + (size_t)idx * TT + base + hi * 8;
  vf.h[0] = *(const v8h*)(vr);
  vf.h[1] = *(const v8h*)(vr + 16);

  v8f S0 = wmma16x16x32(qa.v, kf0.v, zero8f());  // keys base..base+15
  v8f S1 = wmma16x16x32(qa.v, kf1.v, zero8f());  // keys base+16..base+31

  float s0v[8], s1v[8];
  float local = -3.0e38f;
#pragma unroll
  for (int r = 0; r < 8; ++r) {
    if (MASK) {
      int i = ibase + r;
      s0v[r] = (base + idx <= i) ? S0[r] : -3.0e38f;
      s1v[r] = (base + 16 + idx <= i) ? S1[r] : -3.0e38f;
    } else {
      s0v[r] = S0[r];
      s1v[r] = S1[r];
    }
    local = fmaxf(local, fmaxf(s0v[r], s1v[r]));
  }
  float mnew = fmaxf(mtile, redmax32(local));
  float sc = __expf(mtile - mnew);
  mtile = mnew;
#pragma unroll
  for (int r = 0; r < 8; ++r) {
    float e0 = __expf(s0v[r] - mnew);
    float e1 = __expf(s1v[r] - mnew);
    O[r] *= sc;  // rescale accumulator (incl. denominator column 10)
    lds_p[(hi * 8 + r) * 32 + idx] = (_Float16)e0;
    lds_p[(hi * 8 + r) * 32 + 16 + idx] = (_Float16)e1;
  }
  lds_fence();  // same-wave DS in-order; prevent compiler reordering only

  AF pf;
  pf.h[0] = *(const v8h*)(lds_p + idx * 32 + hi * 8);
  pf.h[1] = *(const v8h*)(lds_p + idx * 32 + 16 + hi * 8);
  O = wmma16x16x32(pf.v, vf.v, O);
  lds_fence();  // reads done before next iteration's stores
}

__global__ __launch_bounds__(32) void attn_kernel(
    const _Float16* __restrict__ Q, const _Float16* __restrict__ K,
    const _Float16* __restrict__ Vt, _Float16* __restrict__ attnP) {
  __shared__ __align__(16) _Float16 lds_p[16 * 32];

  const int QT = TT / 16;
  int bid = blockIdx.x;
  int qt = bid % QT;
  int bh = bid / QT;
  int h = bh % NH;
  int b = bh / NH;

  int lane = threadIdx.x;
  int idx = lane & 15;
  int hi = lane >> 4;

  const _Float16* Qb = Q + (size_t)bh * TT * HSP;
  const _Float16* Kb = K + (size_t)bh * TT * HSP;
  const _Float16* Vb = Vt + (size_t)bh * HSP * TT;

  AF qa;
  qa.h[0] = hi ? zero8h() : *(const v8h*)(Qb + (size_t)(qt * 16 + idx) * HSP);
  qa.h[1] = hi ? zero8h() : *(const v8h*)(Qb + (size_t)(qt * 16 + idx) * HSP + 8);

  v8f O = zero8f();  // col 10 accumulates softmax denominator
  float mtile = -3.0e38f;
  int ibase = qt * 16 + hi * 8;

  int kb_end = (qt * 16 + 15) >> 5;
  int full = (qt * 16 - 31) >= 0 ? ((qt * 16 - 31) >> 5) + 1 : 0;
  int kb = 0;
  for (; kb < full; ++kb) {
    int base = kb * 32;
    __builtin_prefetch(Kb + (size_t)(base + 32 + idx + hi * 16) * HSP, 0, 1);
    __builtin_prefetch(Vb + (size_t)idx * TT + base + 32 + hi * 16, 0, 1);
    attn_block<false>(base, ibase, idx, hi, Kb, Vb, qa, lds_p, O, mtile);
  }
  for (; kb <= kb_end; ++kb) {
    attn_block<true>(kb * 32, ibase, idx, hi, Kb, Vb, qa, lds_p, O, mtile);
  }

#pragma unroll
  for (int r = 0; r < 8; ++r) {
    float l = __shfl(O[r], (hi << 4) + HS);  // denominator from column 10
    float val = O[r] / l;
    int row = qt * 16 + hi * 8 + r;
    int col = h * HS + idx;
    _Float16* arow = attnP + ((size_t)b * TT + row) * DM;
    if (idx < HS) arow[col] = (_Float16)val;
    else if (h == NH - 1 && col < DM) arow[col] = (_Float16)0;  // pad 60..63
  }
}

// ---------------------------------------------------------------------------
// K3 (fused tail): proj + LN1 + residual -> x1 (registers) -> FFN1 + GELU
// -> FFN2 + LN2 + residual -> out.  Intermediates never touch HBM:
//   x1 tile lives in 4x v8f; its f16 copy bounces through a 2 KB/wave LDS
//   slice to become FFN1 A-fragments; each 32-col chunk of the GELU output
//   bounces through a 1 KB/wave LDS slice to become FFN2 A-fragments.
// LDS: 8 + 32 + 32 KB weights + 8*(2+1) KB tiles = 96 KB.
// ---------------------------------------------------------------------------
__global__ __launch_bounds__(256) void tail_kernel(
    const _Float16* __restrict__ attnP, const float* __restrict__ Wp,
    const float* __restrict__ bp, const float* __restrict__ g1,
    const float* __restrict__ bln1, const float* __restrict__ x,
    const float* __restrict__ W1, const float* __restrict__ b1f,
    const float* __restrict__ W2, const float* __restrict__ b2f,
    const float* __restrict__ g2, const float* __restrict__ bln2,
    float* __restrict__ out) {
  __shared__ __align__(16) _Float16 wtp[DM * DM];    // W_proj^T (k>=60 zero)
  __shared__ __align__(16) _Float16 w1t[DFF * DM];   // W1^T
  __shared__ __align__(16) _Float16 w2t[DM * DFF];   // W2^T
  __shared__ __align__(16) _Float16 xh[8][16 * DM];  // per-wave x1h tile
  __shared__ __align__(16) _Float16 pt[8][16 * 32];  // per-wave h-chunk tile

  for (int i = threadIdx.x; i < DM * DM; i += 256) {
    int nn = i >> 6, kk = i & 63;
    wtp[i] = (kk < NH * HS) ? (_Float16)Wp[kk * DM + nn] : (_Float16)0;
  }
  for (int i = threadIdx.x; i < DFF * DM; i += 256) {
    int nn = i >> 6, kk = i & 63;          // w1t[n:256][k:64]
    w1t[i] = (_Float16)W1[kk * DFF + nn];
  }
  for (int i = threadIdx.x; i < DM * DFF; i += 256) {
    int nn = i >> 8, kk = i & 255;         // w2t[n:64][k:256]
    w2t[i] = (_Float16)W2[kk * DM + nn];
  }
  __syncthreads();

  int wave = threadIdx.x >> 5, lane = threadIdx.x & 31;
  int idx = lane & 15, hi = lane >> 4;
  int row0 = (blockIdx.x * 8 + wave) * 16;
  _Float16* xs = xh[wave];
  _Float16* ps = pt[wave];

  // ---- proj: attn[16x60pad64] @ Wp -> x1 = x + LN1(. + bp) --------------
  const _Float16* ar = attnP + (size_t)row0 * DM;
  AF a0, a1;
  a0.h[0] = *(const v8h*)(ar + (size_t)idx * DM + hi * 16);
  a0.h[1] = *(const v8h*)(ar + (size_t)idx * DM + hi * 16 + 8);
  a1.h[0] = *(const v8h*)(ar + (size_t)idx * DM + 32 + hi * 16);
  a1.h[1] = *(const v8h*)(ar + (size_t)idx * DM + 32 + hi * 16 + 8);

  v8f x1r[4];
#pragma unroll
  for (int nt = 0; nt < 4; ++nt) {
    AF b0, b1v;
    const _Float16* wr = wtp + (nt * 16 + idx) * DM;
    b0.h[0]  = *(const v8h*)(wr + hi * 16);
    b0.h[1]  = *(const v8h*)(wr + hi * 16 + 8);
    b1v.h[0] = *(const v8h*)(wr + 32 + hi * 16);
    b1v.h[1] = *(const v8h*)(wr + 32 + hi * 16 + 8);
    v8f c = wmma16x16x32(a0.v, b0.v, zero8f());
    x1r[nt] = wmma16x16x32(a1.v, b1v.v, c);
  }

  float cb[4], cg[4], cbe[4];
#pragma unroll
  for (int nt = 0; nt < 4; ++nt) {
    int col = nt * 16 + idx;
    cb[nt] = bp[col]; cg[nt] = g1[col]; cbe[nt] = bln1[col];
  }
#pragma unroll
  for (int nt = 0; nt < 4; ++nt)
#pragma unroll
    for (int r = 0; r < 8; ++r) x1r[nt][r] += cb[nt];

#pragma unroll
  for (int r = 0; r < 8; ++r) {
    float s = x1r[0][r] + x1r[1][r] + x1r[2][r] + x1r[3][r];
    float mu = redadd16(s) * (1.0f / DM);
    float v2 = 0.0f;
#pragma unroll
    for (int nt = 0; nt < 4; ++nt) { float d = x1r[nt][r] - mu; v2 += d * d; }
    float rstd = rsqrtf(redadd16(v2) * (1.0f / DM) + 1e-5f);
    size_t row = (size_t)row0 + hi * 8 + r;
#pragma unroll
    for (int nt = 0; nt < 4; ++nt) {
      int col = nt * 16 + idx;
      float val = (x1r[nt][r] - mu) * rstd * cg[nt] + cbe[nt] + x[row * DM + col];
      x1r[nt][r] = val;                                  // keep residual input
      xs[(hi * 8 + r) * DM + col] = (_Float16)val;       // stage for transpose
    }
  }
  lds_fence();

  // ---- FFN1 A-fragments from staged x1h tile ----------------------------
  AF f0, f1;
  f0.h[0] = *(const v8h*)(xs + idx * DM + hi * 16);
  f0.h[1] = *(const v8h*)(xs + idx * DM + hi * 16 + 8);
  f1.h[0] = *(const v8h*)(xs + idx * DM + 32 + hi * 16);
  f1.h[1] = *(const v8h*)(xs + idx * DM + 32 + hi * 16 + 8);

  // ---- FFN1 (+GELU) interleaved with FFN2 over 32-col chunks ------------
  v8f acc2[4];
#pragma unroll
  for (int nt = 0; nt < 4; ++nt) acc2[nt] = zero8f();

  for (int ks = 0; ks < 8; ++ks) {
#pragma unroll
    for (int sub = 0; sub < 2; ++sub) {
      int nt = ks * 2 + sub;
      AF b0, b1v;
      const _Float16* wr = w1t + (size_t)(nt * 16 + idx) * DM;
      b0.h[0]  = *(const v8h*)(wr + hi * 16);
      b0.h[1]  = *(const v8h*)(wr + hi * 16 + 8);
      b1v.h[0] = *(const v8h*)(wr + 32 + hi * 16);
      b1v.h[1] = *(const v8h*)(wr + 32 + hi * 16 + 8);
      v8f c = wmma16x16x32(f0.v, b0.v, zero8f());
      c = wmma16x16x32(f1.v, b1v.v, c);
      float bias = b1f[nt * 16 + idx];
#pragma unroll
      for (int r = 0; r < 8; ++r) {
        float y = c[r] + bias;
        float hv = 0.5f * y * (1.0f + erff(y * 0.7071067811865476f));
        ps[(hi * 8 + r) * 32 + sub * 16 + idx] = (_Float16)hv;
      }
    }
    lds_fence();
    AF af;
    af.h[0] = *(const v8h*)(ps + idx * 32 + hi * 16);
    af.h[1] = *(const v8h*)(ps + idx * 32 + hi * 16 + 8);
#pragma unroll
    for (int nt = 0; nt < 4; ++nt) {
      AF bf;
      const _Float16* wr =
          w2t + (size_t)(nt * 16 + idx) * DFF + ks * 32 + hi * 16;
      bf.h[0] = *(const v8h*)(wr);
      bf.h[1] = *(const v8h*)(wr + 8);
      acc2[nt] = wmma16x16x32(af.v, bf.v, acc2[nt]);
    }
    lds_fence();
  }

  // ---- bias + LN2 + residual (x1 still in registers) --------------------
#pragma unroll
  for (int nt = 0; nt < 4; ++nt) {
    int col = nt * 16 + idx;
    cb[nt] = b2f[col]; cg[nt] = g2[col]; cbe[nt] = bln2[col];
  }
#pragma unroll
  for (int nt = 0; nt < 4; ++nt)
#pragma unroll
    for (int r = 0; r < 8; ++r) acc2[nt][r] += cb[nt];

#pragma unroll
  for (int r = 0; r < 8; ++r) {
    float s = acc2[0][r] + acc2[1][r] + acc2[2][r] + acc2[3][r];
    float mu = redadd16(s) * (1.0f / DM);
    float v2 = 0.0f;
#pragma unroll
    for (int nt = 0; nt < 4; ++nt) { float d = acc2[nt][r] - mu; v2 += d * d; }
    float rstd = rsqrtf(redadd16(v2) * (1.0f / DM) + 1e-5f);
    size_t row = (size_t)row0 + hi * 8 + r;
#pragma unroll
    for (int nt = 0; nt < 4; ++nt) {
      int col = nt * 16 + idx;
      out[row * DM + col] =
          (acc2[nt][r] - mu) * rstd * cg[nt] + cbe[nt] + x1r[nt][r];
    }
  }
}

// ---------------------------------------------------------------------------
extern "C" void kernel_launch(void* const* d_in, const int* in_sizes, int n_in,
                              void* d_out, int out_size, void* d_ws,
                              size_t ws_size, hipStream_t stream) {
  const float* x    = (const float*)d_in[0];
  const float* Wqkv = (const float*)d_in[1];
  const float* Wp   = (const float*)d_in[2];
  const float* bp   = (const float*)d_in[3];
  const float* g1   = (const float*)d_in[4];
  const float* b1n  = (const float*)d_in[5];
  const float* W1   = (const float*)d_in[6];
  const float* b1   = (const float*)d_in[7];
  const float* W2   = (const float*)d_in[8];
  const float* b2   = (const float*)d_in[9];
  const float* g2   = (const float*)d_in[10];
  const float* b2n  = (const float*)d_in[11];
  float* out = (float*)d_out;
  (void)in_sizes; (void)n_in; (void)out_size; (void)ws_size;

  char* ws = (char*)d_ws;
  size_t off = 0;
  auto carve = [&](size_t bytes) -> void* {
    void* p = ws + off;
    off += (bytes + 255) & ~(size_t)255;
    return p;
  };
  const size_t nBT = (size_t)BB * TT;
  _Float16* Q     = (_Float16*)carve((size_t)BB * NH * TT * HSP * 2);
  _Float16* K     = (_Float16*)carve((size_t)BB * NH * TT * HSP * 2);
  _Float16* Vt    = (_Float16*)carve((size_t)BB * NH * TT * HSP * 2);
  _Float16* attnP = (_Float16*)carve(nBT * DM * 2);

  qkv_kernel<<<dim3((unsigned)(nBT / 128)), dim3(256), 0, stream>>>(
      x, Wqkv, Q, K, Vt);
  attn_kernel<<<dim3((unsigned)(BB * NH * (TT / 16))), dim3(32), 0, stream>>>(
      Q, K, Vt, attnP);
  tail_kernel<<<dim3((unsigned)(nBT / 128)), dim3(256), 0, stream>>>(
      attnP, Wp, bp, g1, b1n, x, W1, b1, W2, b2, g2, b2n, out);
}
